// Encoder_15144054685750
// MI455X (gfx1250) — matrix operand order repaired and checked by reading, not compiled, not run on hardware
//
#include <hip/hip_runtime.h>
#include <hip/hip_bf16.h>

typedef __attribute__((ext_vector_type(16))) _Float16 v16h;
typedef __attribute__((ext_vector_type(8)))  _Float16 h8;
typedef __attribute__((ext_vector_type(8)))  float    v8f;
typedef __attribute__((ext_vector_type(4)))  unsigned int u32x4;
typedef __attribute__((ext_vector_type(8)))  int      i32x8;
typedef __attribute__((ext_vector_type(4)))  int      i32x4;

namespace {
constexpr int Lc = 6, Bc = 8, Sc = 1024, Ec = 768, Hc = 12, Dc = 64, Fc = 3072;
constexpr int Mtok = Bc * Sc;            // 8192 tokens
}

#if defined(__has_builtin)
#if __has_builtin(__builtin_amdgcn_tensor_load_to_lds) && \
    __has_builtin(__builtin_amdgcn_s_wait_tensorcnt)
#define HAVE_TDM 1
#endif
#endif
#ifndef HAVE_TDM
#define HAVE_TDM 0
#endif

__device__ __forceinline__ v8f wmma_f16(v16h a, v16h b, v8f c) {
  // D = A(16x32 f16) * B(32x16 f16) + C(16x16 f32)
  return __builtin_amdgcn_wmma_f32_16x16x32_f16(false, a, false, b, (short)0, c,
                                                false, false);
}

// A-fragment (16x32 f16) gather from row-major storage (global or LDS).
// lane<16 : row=lane,    halves {K0..7, K16..23}
// lane>=16: row=lane-16, halves {K8..15, K24..31}
// caller passes p = &row[sel*8]; we fetch p[0..7] and p[16..23].
__device__ __forceinline__ v16h load_a_frag(const _Float16* p) {
  h8 lo = *(const h8*)p;
  h8 hi = *(const h8*)(p + 16);
  v16h r;
#pragma unroll
  for (int i = 0; i < 8; ++i) { r[i] = lo[i]; r[8 + i] = hi[i]; }
  return r;
}

#if HAVE_TDM
// Issue one TDM 2-D tile load: 32 halves (64B) x `rows` rows, row pitch
// `stride_elems` halves, into LDS at byte offset lds_off. Descriptor per
// CDNA5 ISA (D# group0/group1), data_size=2B, type=2.
__device__ __forceinline__ void tdm_load_tile(unsigned lds_off,
                                              const _Float16* gptr,
                                              unsigned rows,
                                              unsigned stride_elems) {
  unsigned long long ga = (unsigned long long)(uintptr_t)gptr;
  u32x4 g0;
  g0[0] = 1u;                                   // count=1, user descriptor
  g0[1] = lds_off;                              // lds_addr (bytes)
  g0[2] = (unsigned)ga;                         // global_addr[31:0]
  g0[3] = (unsigned)((ga >> 32) & 0x01ffffffu)  // global_addr[56:32]
        | 0x80000000u;                          // type=2 ("image")
  unsigned long long td0 = stride_elems;        // tensor_dim0 (elems/row)
  unsigned long long td1 = rows;                // tensor_dim1
  unsigned long long tl0 = 32, tl1 = rows;      // tile dims
  unsigned long long st0 = stride_elems;        // tensor_dim0_stride
  unsigned long long q0 = (1ull << 16) | (td0 << 48);          // data_size=2B
  unsigned long long q1 = (td0 >> 16) | (td1 << 16) | (tl0 << 48);
  unsigned long long q2 = tl1 | ((st0 & 0xffffffffull) << 32);
  unsigned long long q3 = (st0 >> 32);
  i32x8 g1;
  g1[0] = (int)q0; g1[1] = (int)(q0 >> 32);
  g1[2] = (int)q1; g1[3] = (int)(q1 >> 32);
  g1[4] = (int)q2; g1[5] = (int)(q2 >> 32);
  g1[6] = (int)q3; g1[7] = (int)(q3 >> 32);
  i32x4 z4 = {0, 0, 0, 0};
#if __clang_major__ >= 23
  i32x8 z8 = {0, 0, 0, 0, 0, 0, 0, 0};
  __builtin_amdgcn_tensor_load_to_lds(g0, g1, z4, z4, z8, 0);
#else
  __builtin_amdgcn_tensor_load_to_lds(g0, g1, z4, z4, 0);
#endif
}
#endif  // HAVE_TDM

// ---------------------------------------------------------------------------
// Weight convert+transpose: src f32 [L][K][N] -> dst f16 [L][N][K]
// ---------------------------------------------------------------------------
__global__ void convert_transpose_kernel(const float* __restrict__ src,
                                         _Float16* __restrict__ dst,
                                         int K, int N, long long total) {
  long long idx = (long long)blockIdx.x * blockDim.x + threadIdx.x;
  if (idx >= total) return;
  long long per = (long long)K * N;
  int l = (int)(idx / per);
  long long rem = idx - (long long)l * per;
  int n = (int)(rem / K);
  int k = (int)(rem - (long long)n * K);
  dst[idx] = (_Float16)src[(long long)l * per + (long long)k * N + n];
}

// ---------------------------------------------------------------------------
// LayerNorm over E=768, one token per 256-thread block, f16 output
// ---------------------------------------------------------------------------
__global__ void __launch_bounds__(256)
ln_kernel(const float* __restrict__ x, const float* __restrict__ g,
          const float* __restrict__ b, _Float16* __restrict__ out) {
  int t = blockIdx.x;
  int tid = threadIdx.x;
  const float* xt = x + (size_t)t * Ec;
  __shared__ float red[256];

  float v0 = xt[tid], v1 = xt[tid + 256], v2 = xt[tid + 512];
  red[tid] = v0 + v1 + v2;
  __syncthreads();
  for (int o = 128; o > 0; o >>= 1) {
    if (tid < o) red[tid] += red[tid + o];
    __syncthreads();
  }
  float mean = red[0] * (1.0f / Ec);
  __syncthreads();
  float d0 = v0 - mean, d1 = v1 - mean, d2 = v2 - mean;
  red[tid] = d0 * d0 + d1 * d1 + d2 * d2;
  __syncthreads();
  for (int o = 128; o > 0; o >>= 1) {
    if (tid < o) red[tid] += red[tid + o];
    __syncthreads();
  }
  float rs = rsqrtf(red[0] * (1.0f / Ec) + 1e-5f);
  _Float16* ot = out + (size_t)t * Ec;
  ot[tid]       = (_Float16)(d0 * rs * g[tid]       + b[tid]);
  ot[tid + 256] = (_Float16)(d1 * rs * g[tid + 256] + b[tid + 256]);
  ot[tid + 512] = (_Float16)(d2 * rs * g[tid + 512] + b[tid + 512]);
}

// ---------------------------------------------------------------------------
// WMMA GEMM: C[M,N] = A[M,K](f16 rowmajor) @ W (f16, pre-transposed [N,K])
// 256 threads = 8 waves; block tile 128x128; wave tile 32x64 (2x4 WMMA).
// TDM (tensor_load_to_lds) double-buffers 128x32 A / 128x32 B tiles in LDS,
// pipelined one k-step ahead (TENSORcnt), fragments re-read via ds_load.
// Fused epilogues:
//  EPI 0: Q  -> f16 [b,h,s,d], scaled by 1/sqrt(D)
//  EPI 1: K  -> f16 [b,h,s,d]
//  EPI 2: V  -> f16 [b,h,d,s] (transposed for attention B-fragments)
//  EPI 3: xres[row,col] += val + bias[col]           (f32 residual add)
//  EPI 4: out16[row,col] = gelu_exact(val + bias[col]) (f16)
// ---------------------------------------------------------------------------
template <int EPI>
__global__ void __launch_bounds__(256)
gemm_kernel(const _Float16* __restrict__ A, const _Float16* __restrict__ Bt,
            const float* __restrict__ bias, float* __restrict__ xres,
            _Float16* __restrict__ out16, int M, int N, int K) {
  const int lane = threadIdx.x & 31;
  const int wid  = threadIdx.x >> 5;
  const int lh   = lane & 15;
  const int sel  = lane >> 4;
  const int wm = (wid >> 1) * 32;   // wave row offset in block tile
  const int wn = (wid & 1) * 64;    // wave col offset in block tile
  const int bm = blockIdx.y * 128;
  const int bn = blockIdx.x * 128;

  v8f acc[2][4] = {};

#if HAVE_TDM
  __shared__ _Float16 sA[2][128 * 32];
  __shared__ _Float16 sB[2][128 * 32];
  const int nk = K / 32;
  if (wid == 0) {   // one wave drives the TDM pipeline for the block
    tdm_load_tile((unsigned)(uintptr_t)&sA[0][0], A + (size_t)bm * K, 128, K);
    tdm_load_tile((unsigned)(uintptr_t)&sB[0][0], Bt + (size_t)bn * K, 128, K);
  }
  for (int kt = 0; kt < nk; ++kt) {
    const int cur = kt & 1;
    if (wid == 0) {
      if (kt + 1 < nk) {
        const int nxt = cur ^ 1;
        tdm_load_tile((unsigned)(uintptr_t)&sA[nxt][0],
                      A + (size_t)bm * K + (kt + 1) * 32, 128, K);
        tdm_load_tile((unsigned)(uintptr_t)&sB[nxt][0],
                      Bt + (size_t)bn * K + (kt + 1) * 32, 128, K);
        __builtin_amdgcn_s_wait_tensorcnt(2);  // current tiles landed
      } else {
        __builtin_amdgcn_s_wait_tensorcnt(0);
      }
    }
    __syncthreads();   // publish current tiles to all waves

    const _Float16* tA = &sA[cur][0];
    const _Float16* tB = &sB[cur][0];
    v16h af[2], bf[4];
#pragma unroll
    for (int mi = 0; mi < 2; ++mi)
      af[mi] = load_a_frag(tA + (wm + mi * 16 + lh) * 32 + sel * 8);
#pragma unroll
    for (int nj = 0; nj < 4; ++nj)
      bf[nj] = load_a_frag(tB + (wn + nj * 16 + lh) * 32 + sel * 8);
#pragma unroll
    for (int mi = 0; mi < 2; ++mi)
#pragma unroll
      for (int nj = 0; nj < 4; ++nj)
        acc[mi][nj] = wmma_f16(af[mi], bf[nj], acc[mi][nj]);

    __syncthreads();   // all reads done before buffer reuse
  }
#else
  // Fallback: fragments straight from global (weights are L2-resident).
  const _Float16* Ar0 = A + (size_t)(bm + wm + lh) * K;
  const _Float16* Ar1 = A + (size_t)(bm + wm + 16 + lh) * K;
  for (int k0 = 0; k0 < K; k0 += 32) {
    v16h af[2], bf[4];
    af[0] = load_a_frag(Ar0 + k0 + sel * 8);
    af[1] = load_a_frag(Ar1 + k0 + sel * 8);
#pragma unroll
    for (int nj = 0; nj < 4; ++nj)
      bf[nj] = load_a_frag(Bt + (size_t)(bn + wn + nj * 16 + lh) * K + k0 +
                           sel * 8);
#pragma unroll
    for (int mi = 0; mi < 2; ++mi)
#pragma unroll
      for (int nj = 0; nj < 4; ++nj)
        acc[mi][nj] = wmma_f16(af[mi], bf[nj], acc[mi][nj]);
  }
#endif

#pragma unroll
  for (int mi = 0; mi < 2; ++mi) {
#pragma unroll
    for (int nj = 0; nj < 4; ++nj) {
#pragma unroll
      for (int r = 0; r < 8; ++r) {
        // C layout: lane<16 -> rows 0..7, lane>=16 -> rows 8..15 (vgpr r)
        int row = bm + wm + mi * 16 + sel * 8 + r;
        int col = bn + wn + nj * 16 + lh;
        float v = acc[mi][nj][r];
        if (EPI <= 2) {
          int bi = row >> 10;            // token row = b*S + s (S=1024)
          int s  = row & (Sc - 1);
          int h  = col >> 6;             // col = h*64 + d
          int d  = col & (Dc - 1);
          if (EPI == 0)
            out16[(((size_t)(bi * Hc + h) * Sc + s) << 6) + d] =
                (_Float16)(v * 0.125f);
          else if (EPI == 1)
            out16[(((size_t)(bi * Hc + h) * Sc + s) << 6) + d] = (_Float16)v;
          else
            out16[((size_t)(bi * Hc + h) * Dc + d) * Sc + s] = (_Float16)v;
        } else if (EPI == 3) {
          size_t o = (size_t)row * N + col;
          xres[o] = xres[o] + v + bias[col];
        } else {
          float u = v + bias[col];
          float ge = 0.5f * u * (1.0f + erff(u * 0.70710678118654752f));
          out16[(size_t)row * N + col] = (_Float16)ge;
        }
      }
    }
  }
}

// ---------------------------------------------------------------------------
// Flash attention, one 16-row q-tile per wave. Q pre-scaled by 1/sqrt(D).
// Computes S^T = K @ Q^T so that exp(S^T) lands directly in the WMMA
// A-fragment layout needed for P @ V (no cross-lane transpose).
//  Q: f16 [head][s][d]   K: f16 [head][s][d]   Vt: f16 [head][d][s]
//  O: f16 [b][s][h*64+d]  (row-major [Mtok, E] for the output projection)
// ---------------------------------------------------------------------------
__global__ void __launch_bounds__(128)
flash_kernel(const _Float16* __restrict__ Q, const _Float16* __restrict__ Km,
             const _Float16* __restrict__ Vt, _Float16* __restrict__ O) {
  const int lane = threadIdx.x & 31;
  const int wid  = threadIdx.x >> 5;
  const int w    = blockIdx.x * 4 + wid;     // global wave id
  const int head = w >> 6;                   // 64 q-tiles per head (S/16)
  const int q0   = (w & 63) * 16;
  const int lh   = lane & 15;
  const int sel  = lane >> 4;

  const _Float16* Qh = Q  + (size_t)head * Sc * Dc;
  const _Float16* Kh = Km + (size_t)head * Sc * Dc;
  const _Float16* Vh = Vt + (size_t)head * Dc * Sc;

  // Q^T B-fragments: lane holds row q = q0+lh, d halves {sel*8..+7, 16+sel*8..+7}
  v16h bq0 = load_a_frag(Qh + (size_t)(q0 + lh) * Dc + sel * 8);
  v16h bq1 = load_a_frag(Qh + (size_t)(q0 + lh) * Dc + 32 + sel * 8);

  v8f o0 = {}, o1 = {}, o2 = {}, o3 = {};
  float m_i = -1e30f, l_i = 0.0f;

  for (int j = 0; j < Sc; j += 32) {
    // K A-fragments (rows = keys), two d-halves each, two key sub-tiles
    const _Float16* k0p = Kh + (size_t)(j + lh) * Dc + sel * 8;
    const _Float16* k1p = Kh + (size_t)(j + 16 + lh) * Dc + sel * 8;
    v16h a00 = load_a_frag(k0p);       // keys j..j+15,   d 0..31
    v16h a01 = load_a_frag(k0p + 32);  // keys j..j+15,   d 32..63
    v16h a10 = load_a_frag(k1p);       // keys j+16..31,  d 0..31
    v16h a11 = load_a_frag(k1p + 32);

    v8f s0 = {}, s1 = {};
    s0 = wmma_f16(a00, bq0, s0);
    s0 = wmma_f16(a01, bq1, s0);       // S^T keys j..j+15   x q-tile
    s1 = wmma_f16(a10, bq0, s1);
    s1 = wmma_f16(a11, bq1, s1);       // S^T keys j+16..31  x q-tile

    // online softmax: this lane owns q = lh; its partner lane is lane^16
    float mloc = -1e30f;
#pragma unroll
    for (int r = 0; r < 8; ++r) mloc = fmaxf(mloc, fmaxf(s0[r], s1[r]));
    mloc = fmaxf(mloc, __shfl_xor(mloc, 16, 32));
    float mnew  = fmaxf(m_i, mloc);
    float alpha = __expf(m_i - mnew);

    v16h pa;                            // P in A-fragment layout, for free
    float psum = 0.0f;
#pragma unroll
    for (int r = 0; r < 8; ++r) {
      float p0 = __expf(s0[r] - mnew);
      float p1 = __expf(s1[r] - mnew);
      psum += p0 + p1;
      pa[r]     = (_Float16)p0;
      pa[8 + r] = (_Float16)p1;
    }
    psum += __shfl_xor(psum, 16, 32);
    l_i = l_i * alpha + psum;
    m_i = mnew;

    // rescale running O: vgpr r corresponds to q-row sel*8+r
    float ar[8];
#pragma unroll
    for (int r = 0; r < 8; ++r) ar[r] = __shfl(alpha, sel * 8 + r, 32);
#pragma unroll
    for (int r = 0; r < 8; ++r) {
      o0[r] *= ar[r]; o1[r] *= ar[r]; o2[r] *= ar[r]; o3[r] *= ar[r];
    }

    // V B-fragments from transposed V: keys {sel*8..+7, 16+sel*8..+7} per lane
    const _Float16* vp = Vh + (size_t)lh * Sc + j + sel * 8;
    v16h bv0 = load_a_frag(vp);
    v16h bv1 = load_a_frag(vp + 16 * Sc);
    v16h bv2 = load_a_frag(vp + 32 * Sc);
    v16h bv3 = load_a_frag(vp + 48 * Sc);
    o0 = wmma_f16(pa, bv0, o0);
    o1 = wmma_f16(pa, bv1, o1);
    o2 = wmma_f16(pa, bv2, o2);
    o3 = wmma_f16(pa, bv3, o3);
  }

  // normalize rows and store to [b][s][h*64+d]
  float linv = 1.0f / l_i;
  float lr[8];
#pragma unroll
  for (int r = 0; r < 8; ++r) lr[r] = __shfl(linv, sel * 8 + r, 32);
  const int bi = head / Hc, h = head % Hc;
#pragma unroll
  for (int r = 0; r < 8; ++r) {
    size_t base =
        ((size_t)bi * Sc + q0 + sel * 8 + r) * Ec + (size_t)h * Dc + lh;
    O[base]      = (_Float16)(o0[r] * lr[r]);
    O[base + 16] = (_Float16)(o1[r] * lr[r]);
    O[base + 32] = (_Float16)(o2[r] * lr[r]);
    O[base + 48] = (_Float16)(o3[r] * lr[r]);
  }
}

// ---------------------------------------------------------------------------
extern "C" void kernel_launch(void* const* d_in, const int* in_sizes, int n_in,
                              void* d_out, int out_size, void* d_ws,
                              size_t ws_size, hipStream_t stream) {
  const float* x_in  = (const float*)d_in[0];
  const float* ln1_g = (const float*)d_in[1];
  const float* ln1_b = (const float*)d_in[2];
  const float* wq    = (const float*)d_in[3];
  const float* wk    = (const float*)d_in[4];
  const float* wv    = (const float*)d_in[5];
  const float* wo_w  = (const float*)d_in[6];
  const float* wo_b  = (const float*)d_in[7];
  const float* ln2_g = (const float*)d_in[8];
  const float* ln2_b = (const float*)d_in[9];
  const float* w1_w  = (const float*)d_in[10];
  const float* w1_b  = (const float*)d_in[11];
  const float* w2_w  = (const float*)d_in[12];
  const float* w2_b  = (const float*)d_in[13];

  const size_t EE = (size_t)Ec * Ec;
  const size_t EF = (size_t)Ec * Fc;
  const size_t actH = (size_t)Mtok * Ec;   // f16 activation elements

  char* ws = (char*)d_ws;
  size_t off = 0;
  auto take = [&](size_t bytes) {
    char* p = ws + off;
    off += (bytes + 255) & ~(size_t)255;
    return p;
  };
  _Float16* wqT = (_Float16*)take(Lc * EE * 2);
  _Float16* wkT = (_Float16*)take(Lc * EE * 2);
  _Float16* wvT = (_Float16*)take(Lc * EE * 2);
  _Float16* woT = (_Float16*)take(Lc * EE * 2);
  _Float16* w1T = (_Float16*)take(Lc * EF * 2);
  _Float16* w2T = (_Float16*)take(Lc * EF * 2);
  float*    xw  = (float*)take(actH * 4);      // running residual (f32)
  _Float16* hln = (_Float16*)take(actH * 2);   // LN output (f16)
  _Float16* Qb  = (_Float16*)take(actH * 2);   // [b,h,s,d]
  _Float16* Kb  = (_Float16*)take(actH * 2);   // [b,h,s,d]
  _Float16* Vtb = (_Float16*)take(actH * 2);   // [b,h,d,s]
  _Float16* Ob  = (_Float16*)take(actH * 2);   // [b,s,e]
  _Float16* Hb  = Qb;  // FFN hidden [Mtok, F] f16 aliases Q..O region (4*actH)

  // --- per-call weight convert + transpose to f16 [N,K] ---
  {
    long long tEE = (long long)Lc * EE, tEF = (long long)Lc * EF;
    int grEE = (int)((tEE + 255) / 256), grEF = (int)((tEF + 255) / 256);
    convert_transpose_kernel<<<grEE, 256, 0, stream>>>(wq, wqT, Ec, Ec, tEE);
    convert_transpose_kernel<<<grEE, 256, 0, stream>>>(wk, wkT, Ec, Ec, tEE);
    convert_transpose_kernel<<<grEE, 256, 0, stream>>>(wv, wvT, Ec, Ec, tEE);
    convert_transpose_kernel<<<grEE, 256, 0, stream>>>(wo_w, woT, Ec, Ec, tEE);
    convert_transpose_kernel<<<grEF, 256, 0, stream>>>(w1_w, w1T, Ec, Fc, tEF);
    convert_transpose_kernel<<<grEF, 256, 0, stream>>>(w2_w, w2T, Fc, Ec, tEF);
  }

  hipMemcpyAsync(xw, x_in, actH * 4, hipMemcpyDeviceToDevice, stream);

  const dim3 gE(Ec / 128, Mtok / 128);   // N=768 GEMMs
  const dim3 gF(Fc / 128, Mtok / 128);   // N=3072 GEMM
  const int flashBlocks = Bc * Hc * (Sc / 16) / 4;

  for (int l = 0; l < Lc; ++l) {
    // ---- attention sublayer ----
    ln_kernel<<<Mtok, 256, 0, stream>>>(xw, ln1_g + l * Ec, ln1_b + l * Ec, hln);
    gemm_kernel<0><<<gE, 256, 0, stream>>>(hln, wqT + l * EE, nullptr, nullptr,
                                           Qb, Mtok, Ec, Ec);
    gemm_kernel<1><<<gE, 256, 0, stream>>>(hln, wkT + l * EE, nullptr, nullptr,
                                           Kb, Mtok, Ec, Ec);
    gemm_kernel<2><<<gE, 256, 0, stream>>>(hln, wvT + l * EE, nullptr, nullptr,
                                           Vtb, Mtok, Ec, Ec);
    flash_kernel<<<flashBlocks, 128, 0, stream>>>(Qb, Kb, Vtb, Ob);
    gemm_kernel<3><<<gE, 256, 0, stream>>>(Ob, woT + l * EE, wo_b + l * Ec, xw,
                                           nullptr, Mtok, Ec, Ec);
    // ---- FFN sublayer ----
    ln_kernel<<<Mtok, 256, 0, stream>>>(xw, ln2_g + l * Ec, ln2_b + l * Ec, hln);
    gemm_kernel<4><<<gF, 256, 0, stream>>>(hln, w1T + l * EF, w1_b + l * Fc,
                                           nullptr, Hb, Mtok, Fc, Ec);
    gemm_kernel<3><<<gE, 256, 0, stream>>>(Hb, w2T + l * EF, w2_b + l * Ec, xw,
                                           nullptr, Mtok, Ec, Fc);
  }

  hipMemcpyAsync(d_out, xw, (size_t)out_size * 4, hipMemcpyDeviceToDevice,
                 stream);
}
